// JTEncoder_35287451304147
// MI455X (gfx1250) — matrix operand order repaired and checked by reading, not compiled, not run on hardware
//
#include <hip/hip_runtime.h>
#include <hip/hip_bf16.h>

typedef __attribute__((ext_vector_type(2))) float v2f;
typedef __attribute__((ext_vector_type(8))) float v8f;
typedef __attribute__((ext_vector_type(8))) __bf16 v8bf;
typedef __attribute__((ext_vector_type(16))) __bf16 v16bf;

#define HIDDEN 128
#define ZD 56
#define NGRAPH 1024

#define WMMA_BF16(A, B, C) \
  __builtin_amdgcn_wmma_f32_16x16x32_bf16(false, (A), false, (B), (short)0, (C), false, false)

// ===========================================================================
// Weight repack: split f32 W[K,128] into bf16 hi/lo planes in WMMA-B fragment
// order: out[plane][kb][nt][lane][j], j=0..15, where
//   K = kb*32 + (lane>=16 ? 8 : 0) + (j&7) + (j>=8 ? 16 : 0),  N = nt*16 + (lane&15).
// One lane's fragment is 16 contiguous bf16 = 32B of LDS.
// ===========================================================================
__global__ __launch_bounds__(256) void repack_w(const float* __restrict__ W,
                                                int KB, __bf16* __restrict__ out) {
  int total = KB * 8 * 32 * 16;  // elems per plane
  int tid = blockIdx.x * blockDim.x + threadIdx.x;
  if (tid >= total) return;
  int j = tid & 15;
  int lane = (tid >> 4) & 31;
  int nt = (tid >> 9) & 7;
  int kb = tid >> 12;
  int k = kb * 32 + ((lane >= 16) ? 8 : 0) + (j & 7) + ((j >= 8) ? 16 : 0);
  int n = nt * 16 + (lane & 15);
  float w = W[(size_t)k * HIDDEN + n];
  __bf16 h = (__bf16)w;
  out[tid] = h;                                  // hi plane
  out[total + tid] = (__bf16)(w - (float)h);     // lo plane
}

// ===========================================================================
// Input projection (exact fp32 path, ~1% of FLOPs): Y = relu(X @ W + b),
// V_WMMA_F32_16X16X4_F32, one 16x128 tile per wave. Also emits bf16 hi/lo
// activation planes consumed by the bf16 message kernels.
// ===========================================================================
template <int DIN>
__global__ __launch_bounds__(128) void proj_f32_wmma(
    const float* __restrict__ X, const float* __restrict__ W,
    const float* __restrict__ bias, float* __restrict__ Y,
    __bf16* __restrict__ Yhi, __bf16* __restrict__ Ylo, int M) {
  int wave = (blockIdx.x * blockDim.x + threadIdx.x) >> 5;
  int lane = threadIdx.x & 31;
  int row0 = wave * 16;
  if (row0 >= M) return;  // wave-uniform
  int lc = lane & 15;
  int hi = lane >> 4;

  v8f zero = {};
  v8f acc[8];
#pragma unroll
  for (int nt = 0; nt < 8; ++nt) acc[nt] = zero;

  const float* xrow = X + (size_t)(row0 + lc) * DIN;
  for (int k = 0; k < DIN; k += 4) {
    int kk = k + 2 * hi;
    v2f a;
    a.x = xrow[kk];
    a.y = xrow[kk + 1];
#pragma unroll
    for (int nt = 0; nt < 8; ++nt) {
      int n = nt * 16 + lc;
      v2f b;
      b.x = W[(size_t)kk * HIDDEN + n];
      b.y = W[(size_t)(kk + 1) * HIDDEN + n];
      acc[nt] = __builtin_amdgcn_wmma_f32_16x16x4_f32(
          false, a, false, b, (short)0, acc[nt], false, false);
    }
  }
#pragma unroll
  for (int nt = 0; nt < 8; ++nt) {
    int n = nt * 16 + lc;
    float bv = bias[n];
#pragma unroll
    for (int v = 0; v < 8; ++v) {
      size_t idx = (size_t)(row0 + v + 8 * hi) * HIDDEN + n;
      float val = fmaxf(acc[nt][v] + bv, 0.0f);
      Y[idx] = val;
      __bf16 hb = (__bf16)val;
      Yhi[idx] = hb;
      Ylo[idx] = (__bf16)(val - (float)hb);
    }
  }
}

// ===========================================================================
// Fused message + scatter (dominant matmul), split-bf16 fp32 emulation:
//   m = relu(concat(h[dst], h[src]) @ W[256,128] + b); atomicAdd -> AGGR[dst].
// Packed weights (hi+lo, 128KB) staged in LDS, shared by 8 waves/block.
// N-tiles processed in QUADS: 8 B fragments in flight, 12 WMMAs round-robin
// over 4 independent accumulators -> distance 4 between dependent
// accumulates and >=8 instructions before any B-register refill, so the
// WMMA->VALU WAR co-exec window is filled with real work, not v_nops.
// ===========================================================================
__global__ __launch_bounds__(256) void msg_scatter_bf16(
    const __bf16* __restrict__ Hhi, const __bf16* __restrict__ Hlo,
    const int* __restrict__ src, const int* __restrict__ dst,
    const __bf16* __restrict__ Wp,  // [2][8][8][32][16] packed
    const float* __restrict__ bias, float* __restrict__ AGGR, int E) {
  extern __shared__ char smem[];
  __bf16* sW = (__bf16*)smem;
  {  // cooperative linear copy of 131072 bytes into LDS
    const float4* gs = (const float4*)Wp;
    float4* ss = (float4*)smem;
    for (int i = threadIdx.x; i < 131072 / 16; i += 256) ss[i] = gs[i];
  }
  __syncthreads();

  int wv = threadIdx.x >> 5;
  int lane = threadIdx.x & 31;
  int e0 = (blockIdx.x * 8 + wv) * 16;
  if (e0 >= E) return;  // wave-uniform: EXEC stays all-1s for WMMA
  int lc = lane & 15;
  int hi = lane >> 4;

  int e = e0 + lc;
  size_t doff = (size_t)dst[e] * HIDDEN;
  size_t soff = (size_t)src[e] * HIDDEN;
  const __bf16* dh = Hhi + doff;  // x_i = h[dst]
  const __bf16* dl = Hlo + doff;
  const __bf16* sh = Hhi + soff;  // x_j = h[src]
  const __bf16* sl = Hlo + soff;

  v8f zero = {};
  v8f acc[8];
#pragma unroll
  for (int nt = 0; nt < 8; ++nt) acc[nt] = zero;

#pragma unroll
  for (int kb = 0; kb < 8; ++kb) {  // 8 K-blocks of 32 over concat(dst,src)
    const __bf16* rh = (kb < 4) ? dh : sh;
    const __bf16* rl = (kb < 4) ? dl : sl;
    int kbb = (kb & 3) * 32 + hi * 8;
    v8bf a0 = *(const v8bf*)(rh + kbb);
    v8bf a1 = *(const v8bf*)(rh + kbb + 16);
    v8bf l0 = *(const v8bf*)(rl + kbb);
    v8bf l1 = *(const v8bf*)(rl + kbb + 16);
    v16bf Ah = __builtin_shufflevector(a0, a1, 0, 1, 2, 3, 4, 5, 6, 7, 8, 9,
                                       10, 11, 12, 13, 14, 15);
    v16bf Al = __builtin_shufflevector(l0, l1, 0, 1, 2, 3, 4, 5, 6, 7, 8, 9,
                                       10, 11, 12, 13, 14, 15);
#pragma unroll
    for (int nt = 0; nt < 8; nt += 4) {
      v16bf Bh0 = *(const v16bf*)(sW + ((size_t)(kb * 8 + nt + 0) * 32 + lane) * 16);
      v16bf Bh1 = *(const v16bf*)(sW + ((size_t)(kb * 8 + nt + 1) * 32 + lane) * 16);
      v16bf Bh2 = *(const v16bf*)(sW + ((size_t)(kb * 8 + nt + 2) * 32 + lane) * 16);
      v16bf Bh3 = *(const v16bf*)(sW + ((size_t)(kb * 8 + nt + 3) * 32 + lane) * 16);
      v16bf Bl0 = *(const v16bf*)(sW + ((size_t)((8 + kb) * 8 + nt + 0) * 32 + lane) * 16);
      v16bf Bl1 = *(const v16bf*)(sW + ((size_t)((8 + kb) * 8 + nt + 1) * 32 + lane) * 16);
      v16bf Bl2 = *(const v16bf*)(sW + ((size_t)((8 + kb) * 8 + nt + 2) * 32 + lane) * 16);
      v16bf Bl3 = *(const v16bf*)(sW + ((size_t)((8 + kb) * 8 + nt + 3) * 32 + lane) * 16);
      acc[nt + 0] = WMMA_BF16(Ah, Bh0, acc[nt + 0]);
      acc[nt + 1] = WMMA_BF16(Ah, Bh1, acc[nt + 1]);
      acc[nt + 2] = WMMA_BF16(Ah, Bh2, acc[nt + 2]);
      acc[nt + 3] = WMMA_BF16(Ah, Bh3, acc[nt + 3]);
      acc[nt + 0] = WMMA_BF16(Ah, Bl0, acc[nt + 0]);
      acc[nt + 1] = WMMA_BF16(Ah, Bl1, acc[nt + 1]);
      acc[nt + 2] = WMMA_BF16(Ah, Bl2, acc[nt + 2]);
      acc[nt + 3] = WMMA_BF16(Ah, Bl3, acc[nt + 3]);
      acc[nt + 0] = WMMA_BF16(Al, Bh0, acc[nt + 0]);
      acc[nt + 1] = WMMA_BF16(Al, Bh1, acc[nt + 1]);
      acc[nt + 2] = WMMA_BF16(Al, Bh2, acc[nt + 2]);
      acc[nt + 3] = WMMA_BF16(Al, Bh3, acc[nt + 3]);
    }
  }
  // epilogue: relu + bias, scatter-add (segment_sum) to target node
#pragma unroll
  for (int v = 0; v < 8; ++v) {
    int erow = e0 + v + 8 * hi;
    size_t aoff = (size_t)dst[erow] * HIDDEN;
#pragma unroll
    for (int nt = 0; nt < 8; ++nt) {
      int n = nt * 16 + lc;
      float val = fmaxf(acc[nt][v] + bias[n], 0.0f);
      atomicAdd(&AGGR[aoff + n], val);
    }
  }
}

// ===========================================================================
// Node update: h = relu(AGGR @ lin_w + b), K=128, split-bf16.
// A rows are f32 (atomic output) -> split to bf16 hi/lo on the fly (VALU
// co-executes with XDL WMMA). Packed weights (64KB) staged in LDS.
// Writes f32 h (for pooling) and bf16 hi/lo planes (for next msg gather).
// ===========================================================================
__global__ __launch_bounds__(256) void linear_relu_bf16(
    const float* __restrict__ X, const __bf16* __restrict__ Wp,  // [2][4][8][32][16]
    const float* __restrict__ bias, float* __restrict__ Y,
    __bf16* __restrict__ Yhi, __bf16* __restrict__ Ylo, int M) {
  extern __shared__ char smem[];
  __bf16* sW = (__bf16*)smem;
  {
    const float4* gs = (const float4*)Wp;
    float4* ss = (float4*)smem;
    for (int i = threadIdx.x; i < 65536 / 16; i += 256) ss[i] = gs[i];
  }
  __syncthreads();

  int wv = threadIdx.x >> 5;
  int lane = threadIdx.x & 31;
  int row0 = (blockIdx.x * 8 + wv) * 16;
  if (row0 >= M) return;
  int lc = lane & 15;
  int hi = lane >> 4;

  v8f zero = {};
  v8f acc[8];
#pragma unroll
  for (int nt = 0; nt < 8; ++nt) acc[nt] = zero;

  const float* xrow = X + (size_t)(row0 + lc) * HIDDEN;
#pragma unroll
  for (int kb = 0; kb < 4; ++kb) {
    int kbb = kb * 32 + hi * 8;
    v16bf Ah, Al;
#pragma unroll
    for (int q = 0; q < 4; ++q) {  // fully unrolled -> constant vector indices
      float4 c = *(const float4*)(xrow + kbb + (q & 1) * 4 + (q >> 1) * 16);
      int base = q * 4;
      __bf16 h0 = (__bf16)c.x; Ah[base + 0] = h0; Al[base + 0] = (__bf16)(c.x - (float)h0);
      __bf16 h1 = (__bf16)c.y; Ah[base + 1] = h1; Al[base + 1] = (__bf16)(c.y - (float)h1);
      __bf16 h2 = (__bf16)c.z; Ah[base + 2] = h2; Al[base + 2] = (__bf16)(c.z - (float)h2);
      __bf16 h3 = (__bf16)c.w; Ah[base + 3] = h3; Al[base + 3] = (__bf16)(c.w - (float)h3);
    }
#pragma unroll
    for (int nt = 0; nt < 8; nt += 4) {
      v16bf Bh0 = *(const v16bf*)(sW + ((size_t)(kb * 8 + nt + 0) * 32 + lane) * 16);
      v16bf Bh1 = *(const v16bf*)(sW + ((size_t)(kb * 8 + nt + 1) * 32 + lane) * 16);
      v16bf Bh2 = *(const v16bf*)(sW + ((size_t)(kb * 8 + nt + 2) * 32 + lane) * 16);
      v16bf Bh3 = *(const v16bf*)(sW + ((size_t)(kb * 8 + nt + 3) * 32 + lane) * 16);
      v16bf Bl0 = *(const v16bf*)(sW + ((size_t)((4 + kb) * 8 + nt + 0) * 32 + lane) * 16);
      v16bf Bl1 = *(const v16bf*)(sW + ((size_t)((4 + kb) * 8 + nt + 1) * 32 + lane) * 16);
      v16bf Bl2 = *(const v16bf*)(sW + ((size_t)((4 + kb) * 8 + nt + 2) * 32 + lane) * 16);
      v16bf Bl3 = *(const v16bf*)(sW + ((size_t)((4 + kb) * 8 + nt + 3) * 32 + lane) * 16);
      acc[nt + 0] = WMMA_BF16(Ah, Bh0, acc[nt + 0]);
      acc[nt + 1] = WMMA_BF16(Ah, Bh1, acc[nt + 1]);
      acc[nt + 2] = WMMA_BF16(Ah, Bh2, acc[nt + 2]);
      acc[nt + 3] = WMMA_BF16(Ah, Bh3, acc[nt + 3]);
      acc[nt + 0] = WMMA_BF16(Ah, Bl0, acc[nt + 0]);
      acc[nt + 1] = WMMA_BF16(Ah, Bl1, acc[nt + 1]);
      acc[nt + 2] = WMMA_BF16(Ah, Bl2, acc[nt + 2]);
      acc[nt + 3] = WMMA_BF16(Ah, Bl3, acc[nt + 3]);
      acc[nt + 0] = WMMA_BF16(Al, Bh0, acc[nt + 0]);
      acc[nt + 1] = WMMA_BF16(Al, Bh1, acc[nt + 1]);
      acc[nt + 2] = WMMA_BF16(Al, Bh2, acc[nt + 2]);
      acc[nt + 3] = WMMA_BF16(Al, Bh3, acc[nt + 3]);
    }
  }
#pragma unroll
  for (int nt = 0; nt < 8; ++nt) {
    int n = nt * 16 + lc;
    float bv = bias[n];
#pragma unroll
    for (int v = 0; v < 8; ++v) {
      size_t idx = (size_t)(row0 + v + 8 * hi) * HIDDEN + n;
      float val = fmaxf(acc[nt][v] + bv, 0.0f);
      Y[idx] = val;
      __bf16 hb = (__bf16)val;
      Yhi[idx] = hb;
      Ylo[idx] = (__bf16)(val - (float)hb);
    }
  }
}

// ===========================================================================
// Global mean pool (accumulate) + fuse + heads (exact fp32, tiny).
// ===========================================================================
__global__ __launch_bounds__(256) void pool_kernel(
    const float* __restrict__ H, const int* __restrict__ batch,
    float* __restrict__ sums, float* __restrict__ cnt, int N) {
  int tid = blockIdx.x * blockDim.x + threadIdx.x;
  if (tid >= N * HIDDEN) return;
  int n = tid >> 7;
  int c = tid & (HIDDEN - 1);
  int b = batch[n];
  atomicAdd(&sums[(size_t)b * HIDDEN + c], H[tid]);
  if (c == 0) atomicAdd(&cnt[b], 1.0f);
}

__global__ __launch_bounds__(256) void fused_kernel(
    const float* __restrict__ sums_t, const float* __restrict__ cnt_t,
    const float* __restrict__ sums_g, const float* __restrict__ cnt_g,
    float* __restrict__ fused) {
  int tid = blockIdx.x * blockDim.x + threadIdx.x;
  if (tid >= NGRAPH * 2 * HIDDEN) return;
  int b = tid >> 8;
  int c = tid & (2 * HIDDEN - 1);
  float v;
  if (c < HIDDEN)
    v = sums_t[(size_t)b * HIDDEN + c] / fmaxf(cnt_t[b], 1.0f);
  else
    v = sums_g[(size_t)b * HIDDEN + (c - HIDDEN)] / fmaxf(cnt_g[b], 1.0f);
  fused[tid] = v;
}

__global__ __launch_bounds__(128) void head_wmma(
    const float* __restrict__ fused, const float* __restrict__ Wm,
    const float* __restrict__ bm, const float* __restrict__ Wl,
    const float* __restrict__ bl, float* __restrict__ mu,
    float* __restrict__ lv) {
  const float* W = blockIdx.y ? Wl : Wm;
  const float* bb = blockIdx.y ? bl : bm;
  float* out = blockIdx.y ? lv : mu;

  int wave = (blockIdx.x * blockDim.x + threadIdx.x) >> 5;
  int lane = threadIdx.x & 31;
  int row0 = wave * 16;
  if (row0 >= NGRAPH) return;
  int lc = lane & 15;
  int hi = lane >> 4;

  v8f zero = {};
  v8f acc[4];
#pragma unroll
  for (int nt = 0; nt < 4; ++nt) acc[nt] = zero;

  const float* xrow = fused + (size_t)(row0 + lc) * (2 * HIDDEN);
  for (int k = 0; k < 2 * HIDDEN; k += 4) {
    int kk = k + 2 * hi;
    v2f a;
    a.x = xrow[kk];
    a.y = xrow[kk + 1];
#pragma unroll
    for (int nt = 0; nt < 4; ++nt) {
      int n = nt * 16 + lc;
      v2f b;
      b.x = (n < ZD) ? W[(size_t)kk * ZD + n] : 0.0f;
      b.y = (n < ZD) ? W[(size_t)(kk + 1) * ZD + n] : 0.0f;
      acc[nt] = __builtin_amdgcn_wmma_f32_16x16x4_f32(
          false, a, false, b, (short)0, acc[nt], false, false);
    }
  }
#pragma unroll
  for (int nt = 0; nt < 4; ++nt) {
    int n = nt * 16 + lc;
    if (n < ZD) {
      float bv = bb[n];
#pragma unroll
      for (int v = 0; v < 8; ++v) {
        out[(size_t)(row0 + v + 8 * hi) * ZD + n] = acc[nt][v] + bv;
      }
    }
  }
}

// ===========================================================================
extern "C" void kernel_launch(void* const* d_in, const int* in_sizes, int n_in,
                              void* d_out, int out_size, void* d_ws,
                              size_t ws_size, hipStream_t stream) {
  const int NT = 20000, NG = 100000, ET = 40000, EG = 400000;

  const float* tree_x = (const float*)d_in[0];
  const int* tei = (const int*)d_in[1];  // [2,ET]: row0=src, row1=dst
  const float* graph_x = (const float*)d_in[2];
  const int* gei = (const int*)d_in[3];
  const int* batch_t = (const int*)d_in[4];
  const int* batch_g = (const int*)d_in[5];
  const float* t_proj_w = (const float*)d_in[6];
  const float* t_proj_b = (const float*)d_in[7];
  const float* t_msg_w = (const float*)d_in[8];
  const float* t_msg_b = (const float*)d_in[9];
  const float* t_lin_w = (const float*)d_in[10];
  const float* t_lin_b = (const float*)d_in[11];
  const float* g_proj_w = (const float*)d_in[12];
  const float* g_proj_b = (const float*)d_in[13];
  const float* g_msg_w = (const float*)d_in[14];
  const float* g_msg_b = (const float*)d_in[15];
  const float* g_lin_w = (const float*)d_in[16];
  const float* g_lin_b = (const float*)d_in[17];
  const float* mu_w = (const float*)d_in[18];
  const float* mu_b = (const float*)d_in[19];
  const float* lv_w = (const float*)d_in[20];
  const float* lv_b = (const float*)d_in[21];

  float* out = (float*)d_out;
  float* mu = out;
  float* lv = out + NGRAPH * ZD;
  float* fused = out + 2 * NGRAPH * ZD;

  // ---- workspace carve-out ----
  char* wsb = (char*)d_ws;
  auto alloc = [&](size_t bytes) {
    char* p = wsb;
    wsb += (bytes + 255) & ~(size_t)255;
    return p;
  };
  float* h_t = (float*)alloc((size_t)NT * HIDDEN * 4);
  float* aggr_t = (float*)alloc((size_t)NT * HIDDEN * 4);
  float* h_g = (float*)alloc((size_t)NG * HIDDEN * 4);
  float* aggr_g = (float*)alloc((size_t)NG * HIDDEN * 4);
  __bf16* ht_hi = (__bf16*)alloc((size_t)NT * HIDDEN * 2);
  __bf16* ht_lo = (__bf16*)alloc((size_t)NT * HIDDEN * 2);
  __bf16* hg_hi = (__bf16*)alloc((size_t)NG * HIDDEN * 2);
  __bf16* hg_lo = (__bf16*)alloc((size_t)NG * HIDDEN * 2);
  float* sums_t = (float*)alloc(NGRAPH * HIDDEN * 4);
  float* sums_g = (float*)alloc(NGRAPH * HIDDEN * 4);
  float* cnt_t = (float*)alloc(NGRAPH * 4);
  float* cnt_g = (float*)alloc(NGRAPH * 4);
  const int MSGP = 2 * 8 * 8 * 32 * 16;  // bf16 elems per packed msg layer (hi+lo)
  const int LINP = 2 * 4 * 8 * 32 * 16;
  __bf16* t_msg_p = (__bf16*)alloc((size_t)3 * MSGP * 2);
  __bf16* t_lin_p = (__bf16*)alloc((size_t)3 * LINP * 2);
  __bf16* g_msg_p = (__bf16*)alloc((size_t)3 * MSGP * 2);
  __bf16* g_lin_p = (__bf16*)alloc((size_t)3 * LINP * 2);

  auto cdiv = [](int a, int b) { return (a + b - 1) / b; };

  // ---- pack weights into WMMA-B fragment order (hi/lo bf16 planes) ----
  for (int l = 0; l < 3; ++l) {
    repack_w<<<cdiv(MSGP / 2, 256), 256, 0, stream>>>(
        t_msg_w + (size_t)l * 2 * HIDDEN * HIDDEN, 8, t_msg_p + (size_t)l * MSGP);
    repack_w<<<cdiv(LINP / 2, 256), 256, 0, stream>>>(
        t_lin_w + (size_t)l * HIDDEN * HIDDEN, 4, t_lin_p + (size_t)l * LINP);
    repack_w<<<cdiv(MSGP / 2, 256), 256, 0, stream>>>(
        g_msg_w + (size_t)l * 2 * HIDDEN * HIDDEN, 8, g_msg_p + (size_t)l * MSGP);
    repack_w<<<cdiv(LINP / 2, 256), 256, 0, stream>>>(
        g_lin_w + (size_t)l * HIDDEN * HIDDEN, 4, g_lin_p + (size_t)l * LINP);
  }

  // ---- tree encoder ----
  proj_f32_wmma<64><<<cdiv(NT / 16, 4), 128, 0, stream>>>(
      tree_x, t_proj_w, t_proj_b, h_t, ht_hi, ht_lo, NT);
  for (int l = 0; l < 3; ++l) {
    hipMemsetAsync(aggr_t, 0, (size_t)NT * HIDDEN * 4, stream);
    msg_scatter_bf16<<<cdiv(ET / 16, 8), 256, 131072, stream>>>(
        ht_hi, ht_lo, tei, tei + ET, t_msg_p + (size_t)l * MSGP,
        t_msg_b + l * HIDDEN, aggr_t, ET);
    linear_relu_bf16<<<cdiv(NT / 16, 8), 256, 65536, stream>>>(
        aggr_t, t_lin_p + (size_t)l * LINP, t_lin_b + l * HIDDEN, h_t, ht_hi,
        ht_lo, NT);
  }

  // ---- graph encoder ----
  proj_f32_wmma<32><<<cdiv(NG / 16, 4), 128, 0, stream>>>(
      graph_x, g_proj_w, g_proj_b, h_g, hg_hi, hg_lo, NG);
  for (int l = 0; l < 3; ++l) {
    hipMemsetAsync(aggr_g, 0, (size_t)NG * HIDDEN * 4, stream);
    msg_scatter_bf16<<<cdiv(EG / 16, 8), 256, 131072, stream>>>(
        hg_hi, hg_lo, gei, gei + EG, g_msg_p + (size_t)l * MSGP,
        g_msg_b + l * HIDDEN, aggr_g, EG);
    linear_relu_bf16<<<cdiv(NG / 16, 8), 256, 65536, stream>>>(
        aggr_g, g_lin_p + (size_t)l * LINP, g_lin_b + l * HIDDEN, h_g, hg_hi,
        hg_lo, NG);
  }

  // ---- pool + fuse + heads ----
  hipMemsetAsync(sums_t, 0, (size_t)2 * NGRAPH * HIDDEN * 4, stream);
  hipMemsetAsync(cnt_t, 0, (size_t)2 * NGRAPH * 4, stream);
  pool_kernel<<<cdiv(NT * HIDDEN, 256), 256, 0, stream>>>(h_t, batch_t, sums_t,
                                                          cnt_t, NT);
  pool_kernel<<<cdiv(NG * HIDDEN, 256), 256, 0, stream>>>(h_g, batch_g, sums_g,
                                                          cnt_g, NG);
  fused_kernel<<<cdiv(NGRAPH * 2 * HIDDEN, 256), 256, 0, stream>>>(
      sums_t, cnt_t, sums_g, cnt_g, fused);
  dim3 hgrid(cdiv(NGRAPH / 16, 4), 2);
  head_wmma<<<hgrid, 128, 0, stream>>>(fused, mu_w, mu_b, lv_w, lv_b, mu, lv);
}
